// IsgnDecoder_13494787244390
// MI455X (gfx1250) — matrix operand order repaired and checked by reading, not compiled, not run on hardware
//
#include <hip/hip_runtime.h>

typedef __bf16 bf16_t;
typedef __attribute__((ext_vector_type(16))) __bf16 v16bf;
typedef __attribute__((ext_vector_type(8)))  float  v8f;

#define NN    1600
#define NBEAT 200
#define NOTE  128
#define ENC   64
#define OUTD  11
#define MARG  32
#define TREG  32
#define ET    12
#define FIN   299
#define SEC   43
#define FBH   192
#define FINP  304   /* row stride for h/act/hw (19*16) */
#define KP    320   /* padded K for WMMA (10*32)       */
#define CAP   256
#define MT    100   /* 1600/16 */
#define NT    19    /* 304/16  */

__device__ __forceinline__ float sigm(float x){ return 1.f/(1.f+expf(-x)); }

/* ---------------- small setup kernels ---------------- */

__global__ void k_perfz(const float* __restrict__ pv, const float* __restrict__ w,
                        const float* __restrict__ b, float* __restrict__ pz){
  int t = threadIdx.x;
  if(t < ENC){
    float a = b[t];
    for(int k=0;k<16;k++) a += pv[k]*w[k*ENC+t];
    pz[t] = fmaxf(a, 0.f);
  }
}

__global__ void k_init(const float* __restrict__ note, const float* __restrict__ pz,
                       const float* __restrict__ w1, const float* __restrict__ b1,
                       const float* __restrict__ w2, const float* __restrict__ b2,
                       float* __restrict__ h, float* __restrict__ total0){
  __shared__ float sn[NOTE]; __shared__ float hid[ENC]; __shared__ float io[OUTD];
  int j = blockIdx.x, t = threadIdx.x;
  if(t < NOTE) sn[t] = note[j*NOTE + t];
  __syncthreads();
  if(t < ENC){
    float a = b1[t];
    for(int k=0;k<NOTE;k++) a += sn[k]*w1[k*ENC+t];
    hid[t] = fmaxf(a, 0.f);
  }
  __syncthreads();
  if(t < OUTD){
    float a = b2[t];
    for(int k=0;k<ENC;k++) a += hid[k]*w2[k*OUTD+t];
    io[t] = a;
    total0[j*OUTD+t] = a;
  }
  __syncthreads();
  for(int c=t;c<FINP;c+=blockDim.x){
    float v;
    if(c < NOTE)      v = sn[c];
    else if(c < FBH)  v = pz[c-NOTE];
    else if(c < 256)  v = 0.f;           /* beat_hid0 */
    else if(c < 267)  v = io[c-256];     /* init_out  */
    else              v = 0.f;           /* margin0 + pad */
    h[j*FINP + c] = v;
  }
}

/* wa (E,FIN,FIN) -> bf16 transposed [e][n][k], zero-padded to KPxKP */
__global__ void k_wa2bf(const float* __restrict__ wa, bf16_t* __restrict__ wt){
  long idx = (long)blockIdx.x*blockDim.x + threadIdx.x;
  const long tot = (long)ET*KP*KP;
  for(; idx < tot; idx += (long)gridDim.x*blockDim.x){
    int e = (int)(idx/(KP*KP)); int r = (int)(idx%(KP*KP));
    int n = r/KP, k = r%KP;
    float v = 0.f;
    if(n < FIN && k < FIN) v = wa[((long)e*FIN + k)*FIN + n];
    wt[idx] = (bf16_t)v;
  }
}

/* deterministic per-destination edge list build (no atomics) */
__global__ void k_edges(const float* __restrict__ edges, int* __restrict__ cnt,
                        int* __restrict__ eidx, float* __restrict__ ewt){
  int j = blockIdx.x*blockDim.x + threadIdx.x;
  if(j >= NN) return;
  int c = 0;
  for(int e=0;e<ET;e++){
    const float* base = edges + (long)e*NN*NN + j;
    for(int i=0;i<NN;i++){
      float w = base[(long)i*NN];
      if(w != 0.f && c < CAP){
        eidx[j*CAP + c] = (e<<16) | i;
        ewt [j*CAP + c] = w;
        c++;
      }
    }
  }
  cnt[j] = c;
}

/* h (f32, stride FINP) -> bf16 padded (stride KP) */
__global__ void k_h2bf(const float* __restrict__ h, bf16_t* __restrict__ hb){
  int idx = blockIdx.x*blockDim.x + threadIdx.x;
  const int tot = NN*KP;
  for(; idx < tot; idx += gridDim.x*blockDim.x){
    int r = idx/KP, c = idx%KP;
    hb[idx] = (c < FIN) ? (bf16_t)h[r*FINP + c] : (bf16_t)0.f;
  }
}

/* ---------------- WMMA GEMM: hw[e] = Hb(1600xKP) @ WaT[e]^T ---------------- */
/* one wave per 16x16 output tile; K loop of 10x v_wmma_f32_16x16x32_bf16 */
__global__ __launch_bounds__(256) void k_gemm(const bf16_t* __restrict__ hb,
                                              const bf16_t* __restrict__ wt,
                                              float* __restrict__ hw){
  int tile = blockIdx.x*8 + (threadIdx.x >> 5);
  int e  = tile / (MT*NT);
  int r  = tile % (MT*NT);
  int mt = r / NT, nt = r % NT;
  int lane = threadIdx.x & 31;
  int g = lane >> 4, q = lane & 15;
  const bf16_t* arow = hb + (size_t)(mt*16 + q)*KP;                /* A row (M=q+..) */
  const bf16_t* bcol = wt + ((size_t)e*KP + (nt*16 + q))*KP;       /* B col (N-major) */
  v8f acc = {0.f,0.f,0.f,0.f,0.f,0.f,0.f,0.f};
  for(int kk=0; kk<KP; kk+=32){
    union { v16bf v; uint4 u[2]; } A, Bf;
    /* A 16-bit 16x32 layout: lane group g holds K runs [g*8,g*8+8) and [16+g*8,...) */
    A.u[0]  = *(const uint4*)(arow + kk + g*8);
    A.u[1]  = *(const uint4*)(arow + kk + 16 + g*8);
    /* B 16-bit 32x16 layout: lane group g holds K runs [g*8,g*8+8) and [16+g*8,...) */
    Bf.u[0] = *(const uint4*)(bcol + kk + g*8);
    Bf.u[1] = *(const uint4*)(bcol + kk + 16 + g*8);
    acc = __builtin_amdgcn_wmma_f32_16x16x32_bf16(false, A.v, false, Bf.v,
                                                  (short)0, acc, false, false);
  }
  /* C/D layout: VGPR rr -> M = mt*16 + g*8 + rr, N = nt*16 + q */
  int n = nt*16 + q;
  size_t mbase = (size_t)e*NN + (size_t)mt*16 + g*8;
  #pragma unroll
  for(int rr=0; rr<8; rr++)
    hw[(mbase + rr)*FINP + n] = acc[rr];
}

/* sparse aggregation: act[j,s] = ba[s] + sum over incoming edges of w*hw[e,i,s] */
__global__ void k_agg(const float* __restrict__ hw, const int* __restrict__ cnt,
                      const int* __restrict__ eidx, const float* __restrict__ ewt,
                      const float* __restrict__ ba, float* __restrict__ act){
  int j = blockIdx.x, s = threadIdx.x;
  if(s >= FIN) return;
  int c = cnt[j];
  float a = ba[s];
  for(int t=0;t<c;t++){
    int   p = eidx[j*CAP + t];
    float w = ewt [j*CAP + t];
    int e = p >> 16, i = p & 0xFFFF;
    a += w * hw[((size_t)e*NN + i)*FINP + s];
  }
  act[j*FINP + s] = a;
}

/* GRU cell over the SEC slice of h */
__global__ void k_gru(const float* __restrict__ act, float* __restrict__ h,
                      const float* __restrict__ wz, const float* __restrict__ wr,
                      const float* __restrict__ wh, const float* __restrict__ uz,
                      const float* __restrict__ ur, const float* __restrict__ uh){
  __shared__ float sa[FIN]; __shared__ float hs[SEC]; __shared__ float rh[SEC];
  int j = blockIdx.x, t = threadIdx.x;
  for(int c=t;c<FIN;c+=64) sa[c] = act[j*FINP + c];
  if(t < SEC) hs[t] = h[j*FINP + 256 + t];
  __syncthreads();
  float zt = 0.f;
  if(t < SEC){
    float vz=0.f, vr=0.f;
    for(int k=0;k<FIN;k++){ float av=sa[k]; vz += av*wz[k*SEC+t]; vr += av*wr[k*SEC+t]; }
    for(int k=0;k<SEC;k++){ float hv=hs[k]; vz += hv*uz[k*SEC+t]; vr += hv*ur[k*SEC+t]; }
    zt = sigm(vz);
    rh[t] = sigm(vr)*hs[t];
  }
  __syncthreads();
  if(t < SEC){
    float vh=0.f;
    for(int k=0;k<FIN;k++) vh += sa[k]*wh[k*SEC+t];
    for(int k=0;k<SEC;k++) vh += rh[k]*uh[k*SEC+t];
    float cd = tanhf(vh);
    h[j*FINP + 256 + t] = (1.f - zt)*hs[t] + zt*cd;
  }
}

/* segment attention over 8-note beats (generic over D,H,hd<=32/8) */
__global__ void k_att(const float* __restrict__ h, int xoff, int D, int H, int hd,
                      const float* __restrict__ W, const float* __restrict__ b,
                      const float* __restrict__ ctx, float* __restrict__ pool){
  __shared__ float xv[8][MARG]; __shared__ float aa[8][MARG];
  __shared__ float sim[8][8];   __shared__ float wsm[8][8];
  int sb = blockIdx.x, t = threadIdx.x;
  for(int idx=t; idx<8*D; idx+=blockDim.x){
    int tt=idx/D, c=idx%D;
    xv[tt][c] = h[(size_t)(sb*8+tt)*FINP + xoff + c];
  }
  __syncthreads();
  for(int idx=t; idx<8*D; idx+=blockDim.x){
    int tt=idx/D, c=idx%D;
    float v = b[c];
    for(int k=0;k<D;k++) v += xv[tt][k]*W[k*D+c];
    aa[tt][c] = tanhf(v);
  }
  __syncthreads();
  for(int idx=t; idx<8*H; idx+=blockDim.x){
    int tt=idx/H, hh=idx%H;
    float v=0.f;
    for(int d=0;d<hd;d++) v += aa[tt][hh*hd+d]*ctx[hh*hd+d];
    sim[tt][hh]=v;
  }
  __syncthreads();
  if(t < H){
    float m=-1e30f;
    for(int tt=0;tt<8;tt++) m = fmaxf(m, sim[tt][t]);
    float z=0.f;
    for(int tt=0;tt<8;tt++){ float e2=expf(sim[tt][t]-m); wsm[tt][t]=e2; z+=e2; }
    float inv = 1.f/z;
    for(int tt=0;tt<8;tt++) wsm[tt][t] *= inv;
  }
  __syncthreads();
  if(t < D){
    int hh = t/hd;
    float v=0.f;
    for(int tt=0;tt<8;tt++) v += wsm[tt][hh]*xv[tt][t];
    pool[sb*D + t] = v;
  }
}

__global__ void k_bcat(const float* __restrict__ ob, const float* __restrict__ mb,
                       const float* __restrict__ ri, float* __restrict__ bc){
  int idx = blockIdx.x*blockDim.x + threadIdx.x;
  if(idx >= NBEAT*51) return;
  int nb = idx/51, c = idx%51;
  float v;
  if(c < OUTD)            v = ob[nb*OUTD + c];
  else if(c < OUTD+MARG)  v = mb[nb*MARG + (c-OUTD)];
  else                    v = ri[nb*8 + (c-OUTD-MARG)];
  bc[idx] = v;
}

/* one-wave sequential LSTM (one thread per hidden unit) */
__global__ __launch_bounds__(32) void k_lstm(const float* __restrict__ x, int T, int din, int xstride,
                                             const float* __restrict__ wih, const float* __restrict__ whh,
                                             const float* __restrict__ bias,
                                             float* __restrict__ out, int ostride, int ooff, int rev){
  __shared__ float xb[64]; __shared__ float hb[TREG];
  int t = threadIdx.x;
  float c = 0.f;
  hb[t] = 0.f;
  __syncthreads();
  for(int s=0;s<T;s++){
    int tt = rev ? (T-1-s) : s;
    for(int k=t;k<din;k+=32) xb[k] = x[tt*xstride + k];
    __syncthreads();
    float gi=bias[t], gf=bias[TREG+t], gg=bias[2*TREG+t], go=bias[3*TREG+t];
    for(int k=0;k<din;k++){
      float xv = xb[k];
      gi += xv*wih[(size_t)t*din+k];
      gf += xv*wih[(size_t)(TREG+t)*din+k];
      gg += xv*wih[(size_t)(2*TREG+t)*din+k];
      go += xv*wih[(size_t)(3*TREG+t)*din+k];
    }
    for(int k=0;k<TREG;k++){
      float hv = hb[k];
      gi += hv*whh[t*TREG+k];
      gf += hv*whh[(TREG+t)*TREG+k];
      gg += hv*whh[(2*TREG+t)*TREG+k];
      go += hv*whh[(3*TREG+t)*TREG+k];
    }
    c = sigm(gf)*c + sigm(gi)*tanhf(gg);
    float hn = sigm(go)*tanhf(c);
    __syncthreads();
    hb[t] = hn;
    out[tt*ostride + ooff + t] = hn;
    __syncthreads();
  }
}

__global__ void k_tempo(const float* __restrict__ rnn, const float* __restrict__ w,
                        const float* __restrict__ b, float* __restrict__ tmp){
  int t = blockIdx.x*blockDim.x + threadIdx.x;
  if(t >= NBEAT) return;
  float v = b[0];
  for(int k=0;k<64;k++) v += rnn[t*64+k]*w[k];
  tmp[t] = v;
}

__global__ void k_rnnsp(const float* __restrict__ rnn, const int* __restrict__ bn,
                        float* __restrict__ h){
  int idx = blockIdx.x*blockDim.x + threadIdx.x;
  if(idx >= NN*64) return;
  int j = idx/64, c = idx%64;
  h[j*FINP + FBH + c] = rnn[bn[j]*64 + c];
}

__global__ void k_fc(const float* __restrict__ h, const int* __restrict__ bn,
                     const float* __restrict__ tempo,
                     const float* __restrict__ w1, const float* __restrict__ b1,
                     const float* __restrict__ w2, const float* __restrict__ b2,
                     float* __restrict__ finalb, float* __restrict__ outF,
                     float* __restrict__ outT){
  __shared__ float xr[FIN]; __shared__ float hid[MARG];
  int j = blockIdx.x, t = threadIdx.x;
  for(int c=t;c<FIN;c+=64) xr[c] = h[j*FINP + c];
  __syncthreads();
  if(t < MARG){
    float v = b1[t];
    for(int k=0;k<FIN;k++) v += xr[k]*w1[k*MARG+t];
    hid[t] = fmaxf(v, 0.f);
  }
  __syncthreads();
  if(t < OUTD){
    float v;
    if(t == 0) v = tempo[bn[j]];
    else {
      v = b2[t-1];
      for(int k=0;k<MARG;k++) v += hid[k]*w2[k*(OUTD-1) + (t-1)];
    }
    finalb[j*OUTD+t] = v;
    outF[j*OUTD+t]   = v;
    outT[j*OUTD+t]   = v;
  }
}

__global__ void k_hfinal(const float* __restrict__ fb, float* __restrict__ h){
  int idx = blockIdx.x*blockDim.x + threadIdx.x;
  if(idx >= NN*OUTD) return;
  int j = idx/OUTD, c = idx%OUTD;
  h[j*FINP + 256 + c] = fb[idx];
}

/* ---------------- host side ---------------- */

extern "C" void kernel_launch(void* const* d_in, const int* in_sizes, int n_in,
                              void* d_out, int out_size, void* d_ws, size_t ws_size,
                              hipStream_t stream){
  const float* note   = (const float*)d_in[0];
  const float* pvec   = (const float*)d_in[1];
  const float* rinfo  = (const float*)d_in[2];
  const float* edges  = (const float*)d_in[3];
  const int*   bn     = (const int*)  d_in[4];
  const float* wstyle = (const float*)d_in[5];
  const float* bstyle = (const float*)d_in[6];
  const float* wi1    = (const float*)d_in[7];
  const float* bi1    = (const float*)d_in[8];
  const float* wi2    = (const float*)d_in[9];
  const float* bi2    = (const float*)d_in[10];
  const float* gwa    = (const float*)d_in[11];
  const float* gba    = (const float*)d_in[12];
  const float* gwz    = (const float*)d_in[13];
  const float* gwr    = (const float*)d_in[14];
  const float* gwh    = (const float*)d_in[15];
  const float* guz    = (const float*)d_in[16];
  const float* gur    = (const float*)d_in[17];
  const float* guh    = (const float*)d_in[18];
  const float* abw    = (const float*)d_in[19];
  const float* abb    = (const float*)d_in[20];
  const float* abc    = (const float*)d_in[21];
  const float* amw    = (const float*)d_in[22];
  const float* amb    = (const float*)d_in[23];
  const float* amc    = (const float*)d_in[24];
  const float *wih0f=(const float*)d_in[25], *whh0f=(const float*)d_in[26], *lb0f=(const float*)d_in[27];
  const float *wih0b=(const float*)d_in[28], *whh0b=(const float*)d_in[29], *lb0b=(const float*)d_in[30];
  const float *wih1f=(const float*)d_in[31], *whh1f=(const float*)d_in[32], *lb1f=(const float*)d_in[33];
  const float *wih1b=(const float*)d_in[34], *whh1b=(const float*)d_in[35], *lb1b=(const float*)d_in[36];
  const float *wtmp=(const float*)d_in[37], *btmp=(const float*)d_in[38];
  const float *wf1 =(const float*)d_in[39], *bf1 =(const float*)d_in[40];
  const float *wf2 =(const float*)d_in[41], *bf2 =(const float*)d_in[42];
  float* out = (float*)d_out;

  size_t off = 0;
  auto take = [&](size_t bytes)->char*{
    char* p = (char*)d_ws + off;
    off += (bytes + 255) & ~(size_t)255;
    return p;
  };
  float*  h    = (float*) take((size_t)NN*FINP*4);
  float*  act  = (float*) take((size_t)NN*FINP*4);
  float*  hw   = (float*) take((size_t)ET*NN*FINP*4);
  bf16_t* hb   = (bf16_t*)take((size_t)NN*KP*2);
  bf16_t* wt   = (bf16_t*)take((size_t)ET*KP*KP*2);
  int*    cnt  = (int*)   take((size_t)NN*4);
  int*    eidx = (int*)   take((size_t)NN*CAP*4);
  float*  ewt  = (float*) take((size_t)NN*CAP*4);
  float*  pz   = (float*) take((size_t)ENC*4);
  float*  mbp  = (float*) take((size_t)NBEAT*MARG*4);
  float*  obp  = (float*) take((size_t)NBEAT*OUTD*4);
  float*  bc   = (float*) take((size_t)NBEAT*51*4);
  float*  l0   = (float*) take((size_t)NBEAT*64*4);
  float*  l1   = (float*) take((size_t)NBEAT*64*4);
  float*  tmpo = (float*) take((size_t)NBEAT*4);
  float*  fb   = (float*) take((size_t)NN*OUTD*4);

  k_perfz<<<1,64,0,stream>>>(pvec,wstyle,bstyle,pz);
  k_wa2bf<<<1200,256,0,stream>>>(gwa,wt);
  k_edges<<<(NN+255)/256,256,0,stream>>>(edges,cnt,eidx,ewt);
  k_init<<<NN,128,0,stream>>>(note,pz,wi1,bi1,wi2,bi2,h,out+17600);

  for(int seq=0;seq<2;seq++){
    for(int git=0;git<3;git++){
      k_h2bf<<<(NN*KP+255)/256,256,0,stream>>>(h,hb);
      k_gemm<<<(ET*MT*NT)/8,256,0,stream>>>(hb,wt,hw);
      k_agg<<<NN,320,0,stream>>>(hw,cnt,eidx,ewt,gba,act);
      k_gru<<<NN,64,0,stream>>>(act,h,gwz,gwr,gwh,guz,gur,guh);
    }
    k_att<<<NBEAT,64,0,stream>>>(h,267,MARG,8,4,amw,amb,amc,mbp);
    k_att<<<NBEAT,64,0,stream>>>(h,256,OUTD,1,OUTD,abw,abb,abc,obp);
    k_bcat<<<(NBEAT*51+255)/256,256,0,stream>>>(obp,mbp,rinfo,bc);
    k_lstm<<<1,32,0,stream>>>(bc,NBEAT,51,51,wih0f,whh0f,lb0f,l0,64, 0,0);
    k_lstm<<<1,32,0,stream>>>(bc,NBEAT,51,51,wih0b,whh0b,lb0b,l0,64,32,1);
    k_lstm<<<1,32,0,stream>>>(l0,NBEAT,64,64,wih1f,whh1f,lb1f,l1,64, 0,0);
    k_lstm<<<1,32,0,stream>>>(l0,NBEAT,64,64,wih1b,whh1b,lb1b,l1,64,32,1);
    k_tempo<<<1,256,0,stream>>>(l1,wtmp,btmp,tmpo);
    k_rnnsp<<<(NN*64+255)/256,256,0,stream>>>(l1,bn,h);
    k_fc<<<NN,64,0,stream>>>(h,bn,tmpo,wf1,bf1,wf2,bf2,fb,out,out+17600+(1+seq)*17600);
    k_hfinal<<<(NN*OUTD+255)/256,256,0,stream>>>(fb,h);
  }
  (void)in_sizes; (void)n_in; (void)out_size; (void)ws_size;
}
